// ExpertGather_14474039788032
// MI455X (gfx1250) — matrix operand order repaired and checked by reading, not compiled, not run on hardware
//
#include <hip/hip_runtime.h>

// ---------------------------------------------------------------------------
// ExpertGather + per-expert GEMM for MI455X (gfx1250, wave32, WMMA).
//   y[b,e,k,j] = sum_i x[b, Ind[b,e,k], i] * W[e,i,j]
// B=8 T=2048 I=1024 E=16 K=256 J=1024, fp32 in/out.
// f16 operands (data ~N(0,1), well within f16 range), f32 accumulate via
// v_wmma_f32_16x16x32_f16. LDS rows padded to 40 f16 (20 dwords) so all
// fragment ds_load_b128 are bank-conflict-free (L*20 mod 64 covers every
// multiple of 4 once; 4-dword spans tile the 64 banks).
// ---------------------------------------------------------------------------

typedef __attribute__((ext_vector_type(16))) _Float16 v16h;
typedef __attribute__((ext_vector_type(8)))  float    v8f;
typedef __attribute__((ext_vector_type(2)))  _Float16 h2;

#define B_  8
#define T_  2048
#define I_  1024
#define E_  16
#define K_  256
#define J_  1024

#define BM  128   // output rows (k-dim of y) per block
#define BN  128   // output cols (j) per block
#define BK  32    // K-step (matches WMMA f16 K=32)
#define LDST 40   // padded LDS row stride in f16 elements (80 B = 20 dwords)
#define KITERS (I_ / BK)

__device__ __forceinline__ unsigned pkh(float a, float b) {
  h2 v; v.x = (_Float16)a; v.y = (_Float16)b;       // v_cvt_pk_f16_f32
  return __builtin_bit_cast(unsigned, v);
}

union FragU { uint4 u[2]; v16h v; };

__device__ __forceinline__ v16h ld_frag(const _Float16* base, int byte0, int byte1) {
  FragU f;
  f.u[0] = *(const uint4*)((const char*)base + byte0);   // ds_load_b128
  f.u[1] = *(const uint4*)((const char*)base + byte1);   // ds_load_b128
  return f.v;
}

__global__ __launch_bounds__(256) void expert_gather_gemm(
    const float* __restrict__ x, const float* __restrict__ W,
    const long long* __restrict__ Ind, float* __restrict__ y)
{
  // f16 tiles (converted during staging), double-buffered, padded rows.
  // sA: [m][k] (k innermost).  sB: [n][k] (k innermost) so both A and B
  // fragments are two contiguous 16B DS loads per lane.
  __shared__ _Float16 sA[2][BM * LDST];
  __shared__ _Float16 sB[2][BN * LDST];
  __shared__ int sIdx[BM];

  const int t    = threadIdx.x;
  const int blk  = blockIdx.x;
  const int be   = blk >> 4;        // 16 tiles per (b,e)
  const int tile = blk & 15;
  const int mt   = tile >> 3;       // 0..1  (256 / 128)
  const int nt   = tile & 7;        // 0..7  (1024 / 128)
  const int b    = be >> 4;         // E_ = 16
  const int e    = be & 15;
  const int jb   = nt * BN;
  const int m0   = mt * BM;

  if (t < BM)
    sIdx[t] = (int)Ind[((size_t)b * E_ + e) * K_ + m0 + t];
  __syncthreads();

  // ---- staging: global fp32 -> f16 LDS --------------------------------
  auto stageA = [&](int kk, int buf) {
    const int m   = t >> 1;                 // 0..127 local row
    const int hf  = t & 1;                  // which 16-float half of the row
    const int row = sIdx[m];                // gathered token
    const float* src = x + ((size_t)b * T_ + row) * I_ + kk * BK + hf * 16;
    float4 f0 = *(const float4*)(src + 0);
    float4 f1 = *(const float4*)(src + 4);
    float4 f2 = *(const float4*)(src + 8);
    float4 f3 = *(const float4*)(src + 12);
    uint4 lo = make_uint4(pkh(f0.x,f0.y), pkh(f0.z,f0.w), pkh(f1.x,f1.y), pkh(f1.z,f1.w));
    uint4 hi = make_uint4(pkh(f2.x,f2.y), pkh(f2.z,f2.w), pkh(f3.x,f3.y), pkh(f3.z,f3.w));
    uint4* dst = (uint4*)&sA[buf][m * LDST + hf * 16];
    dst[0] = lo; dst[1] = hi;               // 2x ds_store_b128, contiguous
  };

  auto stageB = [&](int kk, int buf) {
    // Thread owns 2 adjacent j-columns x 8 k-rows; packs k-pairs in regs and
    // writes one b128 per column (4-way worst-case bank conflict vs 8-way
    // scatter before).
    const int kg = t >> 6;                  // 0..3
    const int k0 = kg * 8;
    const int jx = t & 63;                  // 0..63
    const int j  = jx * 2;
    const float* src = W + ((size_t)e * I_ + (kk * BK + k0)) * J_ + jb + j;
    float2 r[8];
#pragma unroll
    for (int rr = 0; rr < 8; ++rr)
      r[rr] = *(const float2*)(src + (size_t)rr * J_);
    uint4 c0 = make_uint4(pkh(r[0].x, r[1].x), pkh(r[2].x, r[3].x),
                          pkh(r[4].x, r[5].x), pkh(r[6].x, r[7].x));
    uint4 c1 = make_uint4(pkh(r[0].y, r[1].y), pkh(r[2].y, r[3].y),
                          pkh(r[4].y, r[5].y), pkh(r[6].y, r[7].y));
    *(uint4*)&sB[buf][(j + 0) * LDST + k0] = c0;
    *(uint4*)&sB[buf][(j + 1) * LDST + k0] = c1;
  };

  // ---- wave / lane mapping --------------------------------------------
  const int lane = t & 31;
  const int wid  = t >> 5;        // 8 waves
  const int wm   = wid >> 2;      // 0..1 : 64-row slab
  const int wn   = wid & 3;       // 0..3 : 32-col slab
  const int hlf  = lane >> 4;     // lane group (0: lanes 0-15, 1: 16-31)
  const int lr   = lane & 15;

  v8f acc[4][2] = {};

  stageA(0, 0); stageB(0, 0);
  __syncthreads();

  for (int kk = 0; kk < KITERS; ++kk) {
    const int buf = kk & 1;
    if (kk + 1 < KITERS) { stageA(kk + 1, buf ^ 1); stageB(kk + 1, buf ^ 1); }

    // A fragments: 16x32 f16. Lane group 0 holds K0-7 / K16-23,
    // group 1 holds K8-15 / K24-31 (ISA 16-bit A layout).
    v16h afr[4];
#pragma unroll
    for (int mi = 0; mi < 4; ++mi) {
      const int row = wm * 64 + mi * 16 + lr;
      const _Float16* p = &sA[buf][row * LDST];
      afr[mi] = ld_frag(p, hlf * 16, 32 + hlf * 16);
    }
    // B fragments: 32x16 f16. Lane group 0 holds K0-7 / K16-23, group 1 K8-15 / K24-31.
    v16h bfr[2];
#pragma unroll
    for (int ni = 0; ni < 2; ++ni) {
      const int n = wn * 32 + ni * 16 + lr;
      const _Float16* p = &sB[buf][n * LDST];
      bfr[ni] = ld_frag(p, hlf * 16, 32 + hlf * 16);
    }

#pragma unroll
    for (int mi = 0; mi < 4; ++mi)
#pragma unroll
      for (int ni = 0; ni < 2; ++ni)
        acc[mi][ni] = __builtin_amdgcn_wmma_f32_16x16x32_f16(
            false, afr[mi], false, bfr[ni],
            (short)0, acc[mi][ni], false, false);

    __syncthreads();
  }

  // ---- epilogue: C/D layout: VGPR v -> M = v (+8 for lanes 16-31), lane -> N
  const size_t outBase = ((size_t)b * E_ + e) * K_;
#pragma unroll
  for (int mi = 0; mi < 4; ++mi) {
#pragma unroll
    for (int ni = 0; ni < 2; ++ni) {
      const int mrow = m0 + wm * 64 + mi * 16 + hlf * 8;
      const int j    = jb + wn * 32 + ni * 16 + lr;
      float* dst = y + (outBase + mrow) * J_ + j;
#pragma unroll
      for (int v = 0; v < 8; ++v)
        dst[(size_t)v * J_] = acc[mi][ni][v];
    }
  }
}

extern "C" void kernel_launch(void* const* d_in, const int* in_sizes, int n_in,
                              void* d_out, int out_size, void* d_ws, size_t ws_size,
                              hipStream_t stream) {
  const float*     x   = (const float*)d_in[0];
  const float*     W   = (const float*)d_in[1];
  const long long* Ind = (const long long*)d_in[2];
  float*           y   = (float*)d_out;
  const int grid = B_ * E_ * (K_ / BM) * (J_ / BN);   // 8*16*2*8 = 2048
  expert_gather_gemm<<<grid, 256, 0, stream>>>(x, W, Ind, y);
}
